// ResidualBlock_80539226734609
// MI455X (gfx1250) — matrix operand order repaired and checked by reading, not compiled, not run on hardware
//
#include <hip/hip_runtime.h>
#include <hip/hip_bf16.h>

// ---------------------------------------------------------------------------
// Types for WMMA fragments (gfx1250, wave32)
// ---------------------------------------------------------------------------
typedef __attribute__((ext_vector_type(16))) __bf16 v16bf;
typedef __attribute__((ext_vector_type(8)))  float  v8f;
typedef __attribute__((ext_vector_type(8)))  unsigned short u16x8;

struct FragPair { u16x8 a, b; };

static __device__ __forceinline__ v16bf make_frag(u16x8 lo, u16x8 hi) {
    FragPair p{lo, hi};
    return __builtin_bit_cast(v16bf, p);
}

template<int ACT>
static __device__ __forceinline__ float activate(float x) {
    if constexpr (ACT == 1) {           // exact GELU
        return 0.5f * x * (1.0f + erff(x * 0.70710678118654752f));
    } else if constexpr (ACT == 2) {    // sigmoid
        return 1.0f / (1.0f + expf(-x));
    } else {
        return x;
    }
}

// ---------------------------------------------------------------------------
// bf16 WMMA GEMM:  out = epilogue( A[M,K] @ W[K,N] + bias [+ res] )
// Block tile 128x128, 128 threads (4 waves, 2x2 grid of 64x64 wave tiles),
// K-step 32, double-buffered LDS.
//  - A tile (128x32) is fetched with CDNA5 async copies:
//      global_load_async_to_lds_b128  (ASYNCcnt, s_wait_asynccnt fence)
//  - B tile is staged through VGPRs and transposed into n-major LDS with
//    paired k-rows packed into 32-bit words (16x ds_store_b32 / thread).
// ---------------------------------------------------------------------------
constexpr int BM = 128, BN = 128, BK = 32;
constexpr int LDA = 40;   // LDS row stride (halves) for A tile [128][32]
constexpr int LDB = 40;   // LDS row stride (halves) for B tile stored n-major [128][32]

template<int ACT, bool HAS_RES, bool BF16_OUT>
__global__ __launch_bounds__(128)
void gemm_bf16_kernel(const unsigned short* __restrict__ A,   // [M,K] bf16 bits
                      const unsigned short* __restrict__ W,   // [K,N] bf16 bits
                      const float* __restrict__ bias,         // [N]
                      const float* __restrict__ res,          // [M,N] or null
                      float* __restrict__ outF,               // [M,N] or null
                      __hip_bfloat16* __restrict__ outB,      // [M,N] or null
                      int M, int N, int K)
{
    __shared__ unsigned short sA[2][BM * LDA];
    __shared__ unsigned short sB[2][BN * LDB];   // sB[n][k]

    const int tid  = threadIdx.x;
    const int lane = tid & 31;
    const int wave = tid >> 5;     // 0..3
    const int wr   = wave & 1;     // 64-row block
    const int wc   = wave >> 1;    // 64-col block
    const int m0   = blockIdx.y * BM;
    const int n0   = blockIdx.x * BN;
    const int KT   = K / BK;

    // B global->LDS coords: two pair-chunks; kp covers 0..15 (k-rows 2kp,2kp+1)
    const int kp0   = tid >> 4;          // 0..7
    const int bnoff = (tid & 15) * 8;    // 0..120

    // A async copy: 4 chunks of 8 halves (16B); rows 0..127, koff 0,8,16,24
    auto asyncA = [&](int buf, int kt) {
        const size_t kb = (size_t)kt * BK;
        #pragma unroll
        for (int i = 0; i < 4; ++i) {
            const int c   = tid + i * 128;
            const int row = c >> 2;
            const int ko  = (c & 3) * 8;
            const unsigned short* g = A + (size_t)(m0 + row) * K + kb + ko;
            const unsigned ldsb = (unsigned)(size_t)&sA[buf][row * LDA + ko];
            asm volatile("global_load_async_to_lds_b128 %0, %1, off"
                         :: "v"(ldsb), "v"(g) : "memory");
        }
    };

    u16x8 bR[4];
    auto loadGB = [&](int kt) {
        const size_t kb = (size_t)kt * BK;
        bR[0] = *(const u16x8*)(W + (kb + 2 * kp0)      * (size_t)N + n0 + bnoff);
        bR[1] = *(const u16x8*)(W + (kb + 2 * kp0 + 1)  * (size_t)N + n0 + bnoff);
        bR[2] = *(const u16x8*)(W + (kb + 2 * kp0 + 16) * (size_t)N + n0 + bnoff);
        bR[3] = *(const u16x8*)(W + (kb + 2 * kp0 + 17) * (size_t)N + n0 + bnoff);
    };
    auto storeLB = [&](int buf) {
        #pragma unroll
        for (int j = 0; j < 8; ++j) {   // pack (k, k+1) into one dword, n-major
            const unsigned w0 = (unsigned)bR[0][j] | ((unsigned)bR[1][j] << 16);
            const unsigned w1 = (unsigned)bR[2][j] | ((unsigned)bR[3][j] << 16);
            *(unsigned*)&sB[buf][(bnoff + j) * LDB + 2 * kp0]      = w0;
            *(unsigned*)&sB[buf][(bnoff + j) * LDB + 2 * kp0 + 16] = w1;
        }
    };

    v8f acc[4][4];
    #pragma unroll
    for (int r = 0; r < 4; ++r)
        #pragma unroll
        for (int c = 0; c < 4; ++c)
            #pragma unroll
            for (int v = 0; v < 8; ++v) acc[r][c][v] = 0.0f;

    const int mrow  = wr * 64 + (lane & 15);
    const int half  = lane >> 4;            // K-half selector per ISA layout
    const int ncol0 = wc * 64 + (lane & 15);

    asyncA(0, 0);
    loadGB(0);
    storeLB(0);
    asm volatile("s_wait_asynccnt 0x0" ::: "memory");
    __syncthreads();

    for (int kt = 0; kt < KT; ++kt) {
        const int cur = kt & 1;
        if (kt + 1 < KT) { asyncA(cur ^ 1, kt + 1); loadGB(kt + 1); }

        v16bf af[4], bfr[4];
        #pragma unroll
        for (int r = 0; r < 4; ++r) {
            const unsigned short* base = &sA[cur][(mrow + 16 * r) * LDA];
            u16x8 lo = *(const u16x8*)(base + half * 8);       // K = half*8..+8
            u16x8 hi = *(const u16x8*)(base + 16 + half * 8);  // K = 16+half*8..+8
            af[r] = make_frag(lo, hi);
        }
        #pragma unroll
        for (int c = 0; c < 4; ++c) {
            const unsigned short* base = &sB[cur][(ncol0 + 16 * c) * LDB + half * 8];
            u16x8 lo = *(const u16x8*)(base);                  // K = half*8..+8
            u16x8 hi = *(const u16x8*)(base + 16);             // K = 16+half*8..+8
            bfr[c] = make_frag(lo, hi);
        }
        #pragma unroll
        for (int r = 0; r < 4; ++r)
            #pragma unroll
            for (int c = 0; c < 4; ++c)
                acc[r][c] = __builtin_amdgcn_wmma_f32_16x16x32_bf16(
                    false, af[r], false, bfr[c], (short)0, acc[r][c], false, false);

        if (kt + 1 < KT) storeLB(cur ^ 1);
        asm volatile("s_wait_asynccnt 0x0" ::: "memory");
        __syncthreads();
    }

    // Epilogue. C/D layout: VGPR v -> M = v (lanes 0-15) or 8+v (lanes 16-31).
    const int mtop = m0 + wr * 64 + half * 8;
    #pragma unroll
    for (int r = 0; r < 4; ++r) {
        #pragma unroll
        for (int c = 0; c < 4; ++c) {
            const int nn = n0 + wc * 64 + c * 16 + (lane & 15);
            const float bv = bias[nn];
            #pragma unroll
            for (int v = 0; v < 8; ++v) {
                const int mm = mtop + r * 16 + v;
                float val = activate<ACT>(acc[r][c][v] + bv);
                if constexpr (HAS_RES) val += res[(size_t)mm * N + nn];
                if constexpr (BF16_OUT) outB[(size_t)mm * N + nn] = __float2bfloat16(val);
                else                    outF[(size_t)mm * N + nn] = val;
            }
        }
    }
}

// ---------------------------------------------------------------------------
// Elementwise / small kernels
// ---------------------------------------------------------------------------
__global__ void f32_to_bf16_kernel(const float* __restrict__ in,
                                   __hip_bfloat16* __restrict__ out, size_t n) {
    size_t i = (size_t)blockIdx.x * blockDim.x + threadIdx.x;
    if (i < n) out[i] = __float2bfloat16(in[i]);
}

__global__ __launch_bounds__(256)
void rmsnorm_bf16_kernel(const float* __restrict__ x, const float* __restrict__ w,
                         __hip_bfloat16* __restrict__ out, int D) {
    const int row = blockIdx.x;
    const float* xr = x + (size_t)row * D;
    float s = 0.f;
    for (int i = threadIdx.x; i < D; i += 256) { float v = xr[i]; s += v * v; }
    __shared__ float red[256];
    red[threadIdx.x] = s;
    __syncthreads();
    for (int o = 128; o > 0; o >>= 1) {
        if (threadIdx.x < o) red[threadIdx.x] += red[threadIdx.x + o];
        __syncthreads();
    }
    const float inv = 1.0f / sqrtf(red[0] / (float)D + 1e-8f);
    for (int i = threadIdx.x; i < D; i += 256)
        out[(size_t)row * D + i] = __float2bfloat16(w[i] * xr[i] * inv);
}

// causal depthwise conv, K=4, writes fp32 (scan input) + bf16 (gate GEMM input)
__global__ void conv_kernel(const float* __restrict__ xin, const float* __restrict__ cw,
                            const float* __restrict__ cb, float* __restrict__ outF,
                            __hip_bfloat16* __restrict__ outB, int B_, int S_, int H_) {
    size_t idx = (size_t)blockIdx.x * blockDim.x + threadIdx.x;
    size_t total = (size_t)B_ * S_ * H_;
    if (idx >= total) return;
    int h = (int)(idx % H_);
    size_t t = idx / H_;
    int s = (int)(t % S_);
    int b = (int)(t / S_);
    float acc = cb[h];
    #pragma unroll
    for (int k = 0; k < 4; ++k) {
        int si = s - 3 + k;
        if (si >= 0) acc += cw[k * H_ + h] * xin[((size_t)b * S_ + si) * H_ + h];
    }
    outF[idx] = acc;
    outB[idx] = __float2bfloat16(acc);
}

// RG-LRU chunked scan, pass 1: elementwise (a_t, b_t) in place + chunk composition
__global__ void scan1_kernel(float* __restrict__ rg,      // in: r gate, out: a_t
                             float* __restrict__ ig,      // in: i gate, out: b_t
                             const float* __restrict__ xc,
                             const float* __restrict__ a_param,
                             float* __restrict__ chunkA, float* __restrict__ chunkB,
                             int S_, int H_, int NC_, int CH_, int total) {
    int g = blockIdx.x * blockDim.x + threadIdx.x;
    if (g >= total) return;
    int h = g % H_;
    int c = (g / H_) % NC_;
    int b = g / (H_ * NC_);
    float ap = a_param[h];
    float sp = (ap > 20.f) ? ap : log1pf(expf(ap));   // softplus
    float cfac = -8.0f * sp;                          // Griffin C_RGLRU
    float A = 1.f, Bc = 0.f;
    for (int t = 0; t < CH_; ++t) {
        int s = c * CH_ + t;
        size_t idx = ((size_t)b * S_ + s) * H_ + h;
        float r  = rg[idx];
        float iv = ig[idx];
        float xv = xc[idx];
        float la = cfac * r;
        float a  = expf(la);
        float bt = sqrtf(-expm1f(2.f * la)) * iv * xv;
        rg[idx] = a;
        ig[idx] = bt;
        A  = a * A;
        Bc = a * Bc + bt;
    }
    size_t ci = ((size_t)b * NC_ + c) * H_ + h;
    chunkA[ci] = A;
    chunkB[ci] = Bc;
}

// pass 2: exclusive scan of chunk carries per (b,h)
__global__ void scan2_kernel(const float* __restrict__ chunkA,
                             const float* __restrict__ chunkB,
                             float* __restrict__ carry, int H_, int NC_, int total) {
    int g = blockIdx.x * blockDim.x + threadIdx.x;
    if (g >= total) return;
    int h = g % H_;
    int b = g / H_;
    float st = 0.f;
    for (int c = 0; c < NC_; ++c) {
        size_t ci = ((size_t)b * NC_ + c) * H_ + h;
        carry[ci] = st;
        st = chunkA[ci] * st + chunkB[ci];
    }
}

// pass 3: replay within chunk with carry; fuse left*right -> bf16
__global__ void scan3_kernel(const float* __restrict__ a_arr, const float* __restrict__ b_arr,
                             const float* __restrict__ carry, const float* __restrict__ left,
                             __hip_bfloat16* __restrict__ lr,
                             int S_, int H_, int NC_, int CH_, int total) {
    int g = blockIdx.x * blockDim.x + threadIdx.x;
    if (g >= total) return;
    int h = g % H_;
    int c = (g / H_) % NC_;
    int b = g / (H_ * NC_);
    float hs = carry[((size_t)b * NC_ + c) * H_ + h];
    for (int t = 0; t < CH_; ++t) {
        int s = c * CH_ + t;
        size_t idx = ((size_t)b * S_ + s) * H_ + h;
        hs = a_arr[idx] * hs + b_arr[idx];
        lr[idx] = __float2bfloat16(left[idx] * hs);
    }
}

__global__ void gate_mul_kernel(const __hip_bfloat16* __restrict__ u1,
                                const __hip_bfloat16* __restrict__ u2,
                                __hip_bfloat16* __restrict__ o, size_t n) {
    size_t i = (size_t)blockIdx.x * blockDim.x + threadIdx.x;
    if (i < n) o[i] = __float2bfloat16(__bfloat162float(u1[i]) * __bfloat162float(u2[i]));
}

// ---------------------------------------------------------------------------
// Launch
// ---------------------------------------------------------------------------
extern "C" void kernel_launch(void* const* d_in, const int* in_sizes, int n_in,
                              void* d_out, int out_size, void* d_ws, size_t ws_size,
                              hipStream_t stream) {
    (void)in_sizes; (void)n_in; (void)out_size; (void)ws_size;
    const int D = 2048, H = 2048, F = 6144, Bsz = 2, S = 2048;
    const int M = Bsz * S;           // 4096
    const int NC = 16, CH = S / NC;  // 16 chunks of 128

    const float* x       = (const float*)d_in[0];
    const float* norm1_w = (const float*)d_in[1];
    const float* left_W  = (const float*)d_in[2];
    const float* left_b  = (const float*)d_in[3];
    const float* right_W = (const float*)d_in[4];
    const float* right_b = (const float*)d_in[5];
    const float* conv_w  = (const float*)d_in[6];
    const float* conv_b  = (const float*)d_in[7];
    const float* ga_W    = (const float*)d_in[8];
    const float* ga_b    = (const float*)d_in[9];
    const float* gx_W    = (const float*)d_in[10];
    const float* gx_b    = (const float*)d_in[11];
    const float* a_param = (const float*)d_in[12];
    const float* out_W   = (const float*)d_in[13];
    const float* out_b   = (const float*)d_in[14];
    const float* norm2_w = (const float*)d_in[15];
    const float* up1_W   = (const float*)d_in[16];
    const float* up1_b   = (const float*)d_in[17];
    const float* up2_W   = (const float*)d_in[18];
    const float* up2_b   = (const float*)d_in[19];
    const float* down_W  = (const float*)d_in[20];
    const float* down_b  = (const float*)d_in[21];
    float* outp = (float*)d_out;

    char* ws = (char*)d_ws;
    size_t off = 0;
    auto alloc = [&](size_t bytes) -> void* {
        void* p = ws + off;
        off += (bytes + 255) & ~(size_t)255;
        return p;
    };

    __hip_bfloat16* wLeft  = (__hip_bfloat16*)alloc((size_t)D * H * 2);
    __hip_bfloat16* wRight = (__hip_bfloat16*)alloc((size_t)D * H * 2);
    __hip_bfloat16* wGa    = (__hip_bfloat16*)alloc((size_t)H * H * 2);
    __hip_bfloat16* wGx    = (__hip_bfloat16*)alloc((size_t)H * H * 2);
    __hip_bfloat16* wOut   = (__hip_bfloat16*)alloc((size_t)H * D * 2);
    __hip_bfloat16* wUp1   = (__hip_bfloat16*)alloc((size_t)D * F * 2);
    __hip_bfloat16* wUp2   = (__hip_bfloat16*)alloc((size_t)D * F * 2);
    __hip_bfloat16* wDown  = (__hip_bfloat16*)alloc((size_t)F * D * 2);
    __hip_bfloat16* xn_bf  = (__hip_bfloat16*)alloc((size_t)M * D * 2);  // reused as lr_bf
    float*          left_f = (float*)alloc((size_t)M * H * 4);
    float*          bufA   = (float*)alloc((size_t)M * H * 4);           // right_pre, then x1
    float*          xconvF = (float*)alloc((size_t)M * H * 4);
    __hip_bfloat16* xconvB = (__hip_bfloat16*)alloc((size_t)M * H * 2);  // reused as x1n_bf
    float*          rg     = (float*)alloc((size_t)M * H * 4);           // r -> a_t ; reused for h_bf
    float*          ig     = (float*)alloc((size_t)M * H * 4);           // i -> b_t
    float*          chunkA = (float*)alloc((size_t)Bsz * NC * H * 4);
    float*          chunkB = (float*)alloc((size_t)Bsz * NC * H * 4);
    float*          carry  = (float*)alloc((size_t)Bsz * NC * H * 4);
    __hip_bfloat16* u1_bf  = (__hip_bfloat16*)alloc((size_t)M * F * 2);
    __hip_bfloat16* u2_bf  = (__hip_bfloat16*)alloc((size_t)M * F * 2);
    __hip_bfloat16* h_bf   = (__hip_bfloat16*)rg;  // rg+ig (64MB) dead after scan3; holds M*F bf16

    auto cvt = [&](const float* src, __hip_bfloat16* dst, size_t n) {
        f32_to_bf16_kernel<<<(unsigned)((n + 255) / 256), 256, 0, stream>>>(src, dst, n);
    };

    // 1) weights -> bf16
    cvt(left_W,  wLeft,  (size_t)D * H);
    cvt(right_W, wRight, (size_t)D * H);
    cvt(ga_W,    wGa,    (size_t)H * H);
    cvt(gx_W,    wGx,    (size_t)H * H);
    cvt(out_W,   wOut,   (size_t)H * D);
    cvt(up1_W,   wUp1,   (size_t)D * F);
    cvt(up2_W,   wUp2,   (size_t)D * F);
    cvt(down_W,  wDown,  (size_t)F * D);

    // 2) xn = rmsnorm(x) -> bf16
    rmsnorm_bf16_kernel<<<M, 256, 0, stream>>>(x, norm1_w, xn_bf, D);

    const dim3 blk(128);
    const dim3 gH(H / BN, M / BM);   // N = 2048
    const dim3 gF(F / BN, M / BM);   // N = 6144

    // 3) left = gelu(xn @ left_W + b)            4) right_pre = xn @ right_W + b
    gemm_bf16_kernel<1, false, false><<<gH, blk, 0, stream>>>(
        (const unsigned short*)xn_bf, (const unsigned short*)wLeft, left_b,
        nullptr, left_f, nullptr, M, H, D);
    gemm_bf16_kernel<0, false, false><<<gH, blk, 0, stream>>>(
        (const unsigned short*)xn_bf, (const unsigned short*)wRight, right_b,
        nullptr, bufA, nullptr, M, H, D);

    // 5) causal depthwise conv
    {
        size_t total = (size_t)M * H;
        conv_kernel<<<(unsigned)((total + 255) / 256), 256, 0, stream>>>(
            bufA, conv_w, conv_b, xconvF, xconvB, Bsz, S, H);
    }

    // 6) r = sigmoid(xc @ ga_W + b)              7) i = sigmoid(xc @ gx_W + b)
    gemm_bf16_kernel<2, false, false><<<gH, blk, 0, stream>>>(
        (const unsigned short*)xconvB, (const unsigned short*)wGa, ga_b,
        nullptr, rg, nullptr, M, H, H);
    gemm_bf16_kernel<2, false, false><<<gH, blk, 0, stream>>>(
        (const unsigned short*)xconvB, (const unsigned short*)wGx, gx_b,
        nullptr, ig, nullptr, M, H, H);

    // 8-10) RG-LRU chunked scan, fused left*right -> bf16
    {
        int t1 = Bsz * NC * H;
        scan1_kernel<<<(t1 + 255) / 256, 256, 0, stream>>>(
            rg, ig, xconvF, a_param, chunkA, chunkB, S, H, NC, CH, t1);
        int t2 = Bsz * H;
        scan2_kernel<<<(t2 + 255) / 256, 256, 0, stream>>>(
            chunkA, chunkB, carry, H, NC, t2);
        scan3_kernel<<<(t1 + 255) / 256, 256, 0, stream>>>(
            rg, ig, carry, left_f, xn_bf /* lr_bf */, S, H, NC, CH, t1);
    }

    // 11) x1 = lr @ out_W + out_b + x
    gemm_bf16_kernel<0, true, false><<<gH, blk, 0, stream>>>(
        (const unsigned short*)xn_bf, (const unsigned short*)wOut, out_b,
        x, bufA /* x1 */, nullptr, M, D, H);

    // 12) x1n = rmsnorm(x1) -> bf16
    rmsnorm_bf16_kernel<<<M, 256, 0, stream>>>(bufA, norm2_w, xconvB /* x1n */, D);

    // 13/14) u1 = gelu(x1n @ up1_W + b) bf16 ;  u2 = x1n @ up2_W + b bf16
    gemm_bf16_kernel<1, false, true><<<gF, blk, 0, stream>>>(
        (const unsigned short*)xconvB, (const unsigned short*)wUp1, up1_b,
        nullptr, nullptr, u1_bf, M, F, D);
    gemm_bf16_kernel<0, false, true><<<gF, blk, 0, stream>>>(
        (const unsigned short*)xconvB, (const unsigned short*)wUp2, up2_b,
        nullptr, nullptr, u2_bf, M, F, D);

    // 15) h = u1 * u2 -> bf16
    {
        size_t total = (size_t)M * F;
        gate_mul_kernel<<<(unsigned)((total + 255) / 256), 256, 0, stream>>>(
            u1_bf, u2_bf, h_bf, total);
    }

    // 16) out = h @ down_W + down_b + x1
    gemm_bf16_kernel<0, true, false><<<gH, blk, 0, stream>>>(
        (const unsigned short*)h_bf, (const unsigned short*)wDown, down_b,
        bufA /* x1 */, outp, nullptr, M, D, F);
}
